// _TemporalCfCLayer_34892314313370
// MI455X (gfx1250) — matrix operand order repaired and checked by reading, not compiled, not run on hardware
//
#include <hip/hip_runtime.h>
#include <hip/hip_bf16.h>
#include <stdint.h>

// ---------------------------------------------------------------------------
// TemporalCfCLayer for MI455X (gfx1250, wave32, WMMA 16x16x32 bf16)
//
//   1. zero barrier counters, init h (fp32 + bf16) from `hidden`
//   2. transpose+convert all weights fp32 -> bf16, [K,N] -> [N,K]
//   3. LayerNorm -> xn (bf16)
//   4. WMMA GEMM (2x2 register-blocked): cfc_x = xn @ w_in + b_in
//   5. persistent recurrent kernel (32 WGs x 8 waves = 256 waves):
//        512 steps x { bb0 GEMM | bb1 GEMM | 4-head GEMM + CfC gate } with
//        grid-wide sense barriers; all weights L2-resident (192 MB L2).
//   6. WMMA GEMM (2x2 blocked): y = x + cfc_out @ w_out + b_out  (fp32 out)
//   7. copy h_final (fp32) into tail of d_out
// ---------------------------------------------------------------------------

#define B_SZ   32
#define T_SZ   512
#define DM     1024
#define UNITS  1024
#define BBU    2048
#define NBAR   (T_SZ * 3)
#define RB     32      // persistent blocks (must be co-resident)
#define RT     256     // threads per block -> 8 waves

typedef __bf16 bf16_t;
typedef __attribute__((ext_vector_type(8)))  bf16_t v8bf;
typedef __attribute__((ext_vector_type(16))) bf16_t v16bf;
typedef __attribute__((ext_vector_type(8)))  float  v8f;

__device__ __forceinline__ v8f wmma_bf16(v16bf a, v16bf b, v8f c) {
  return __builtin_amdgcn_wmma_f32_16x16x32_bf16(false, a, false, b, (short)0, c,
                                                 false, false);
}

__device__ __forceinline__ v16bf cat16(v8bf lo, v8bf hi) {
  return __builtin_shufflevector(lo, hi, 0, 1, 2, 3, 4, 5, 6, 7,
                                         8, 9, 10, 11, 12, 13, 14, 15);
}

// A fragment (16x32, MxK): lane holds row m=(lane&15); elements 0..7 are
// K = k0 + (lane>>4)*8 .. +7, elements 8..15 are K = k0+16+(lane>>4)*8 .. +7.
__device__ __forceinline__ v16bf load_a_frag(const bf16_t* __restrict__ row,
                                             int k0, int lane) {
  int koff = k0 + ((lane >> 4) << 3);
  v8bf lo = *(const v8bf*)(row + koff);
  v8bf hi = *(const v8bf*)(row + koff + 16);
  return cat16(lo, hi);
}

// B fragment (32x16, KxN) from weights pre-transposed to Wt[N][K] (K contiguous):
// lane holds column n = n0+(lane&15); elements 0..7 are K = k0 + (lane>>4)*8 .. +7,
// elements 8..15 are K = k0+16+(lane>>4)*8 .. +7.
__device__ __forceinline__ v16bf load_b_frag(const bf16_t* __restrict__ Wt,
                                             int K, int n0, int k0, int lane) {
  const bf16_t* p = Wt + (size_t)(n0 + (lane & 15)) * K + (k0 + ((lane >> 4) << 3));
  v8bf lo = *(const v8bf*)p;
  v8bf hi = *(const v8bf*)(p + 16);
  return cat16(lo, hi);
}

// Branch-free tanh: 1 - 2/(e^{2x}+1).  e^{2x}->inf => 1 ; ->0 => -1.
__device__ __forceinline__ float fast_tanh(float x) {
  float e = __expf(2.0f * x);
  return 1.0f - 2.0f / (e + 1.0f);
}

__device__ __forceinline__ float lecun_tanh(float x) {
  return 1.7159f * fast_tanh(0.666f * x);
}

// ---------------------------------------------------------------------------
// Utility kernels
// ---------------------------------------------------------------------------
__global__ void zero_u32_kernel(unsigned* __restrict__ p, int n) {
  int i = blockIdx.x * blockDim.x + threadIdx.x;
  if (i < n) p[i] = 0u;
}

__global__ void init_hidden_kernel(const float* __restrict__ h0,
                                   bf16_t* __restrict__ hbf,
                                   float* __restrict__ hf32, int n) {
  int i = blockIdx.x * blockDim.x + threadIdx.x;
  if (i < n) { float v = h0[i]; hbf[i] = (bf16_t)v; hf32[i] = v; }
}

__global__ void copy_f32_kernel(const float* __restrict__ in,
                                float* __restrict__ out, int n) {
  int i = blockIdx.x * blockDim.x + threadIdx.x;
  if (i < n) out[i] = in[i];
}

// in: [K][N] fp32 row-major  ->  out: [N][K] bf16 row-major
__global__ void transpose_bf16_kernel(const float* __restrict__ in,
                                      bf16_t* __restrict__ out, int K, int N) {
  size_t total = (size_t)K * N;
  for (size_t i = (size_t)blockIdx.x * blockDim.x + threadIdx.x; i < total;
       i += (size_t)gridDim.x * blockDim.x) {
    size_t n = i / (size_t)K, k = i - n * (size_t)K;
    out[i] = (bf16_t)in[k * (size_t)N + n];
  }
}

// ---------------------------------------------------------------------------
// LayerNorm over DM, output bf16. One block (256 thr) per row.
// ---------------------------------------------------------------------------
__global__ void layernorm_bf16_kernel(const float* __restrict__ x,
                                      const float* __restrict__ w,
                                      const float* __restrict__ b,
                                      bf16_t* __restrict__ xn) {
  __shared__ float s_sum[8], s_sq[8];
  int row = blockIdx.x;
  const float* xr = x + (size_t)row * DM;
  float v[4], s = 0.f, s2 = 0.f;
#pragma unroll
  for (int i = 0; i < 4; ++i) {
    v[i] = xr[threadIdx.x + i * 256];
    s += v[i]; s2 += v[i] * v[i];
  }
#pragma unroll
  for (int o = 16; o; o >>= 1) { s += __shfl_xor(s, o); s2 += __shfl_xor(s2, o); }
  int wv = threadIdx.x >> 5, lane = threadIdx.x & 31;
  if (lane == 0) { s_sum[wv] = s; s_sq[wv] = s2; }
  __syncthreads();
  if (threadIdx.x == 0) {
    float a = 0.f, q = 0.f;
    for (int i = 0; i < 8; ++i) { a += s_sum[i]; q += s_sq[i]; }
    s_sum[0] = a; s_sq[0] = q;
  }
  __syncthreads();
  float mu = s_sum[0] * (1.0f / DM);
  float var = s_sq[0] * (1.0f / DM) - mu * mu;
  float rs = rsqrtf(var + 1e-5f);
#pragma unroll
  for (int i = 0; i < 4; ++i) {
    int idx = threadIdx.x + i * 256;
    xn[(size_t)row * DM + idx] = (bf16_t)(((v[i] - mu) * rs) * w[idx] + b[idx]);
  }
}

// ---------------------------------------------------------------------------
// Big WMMA GEMM, 2x2 register-blocked (32x32 output per wave).
// Each A frag feeds 2 N-tiles, each B frag feeds 2 M-tiles: 2 loads/WMMA.
//   RESID=true : fp32 out = A@Bt^T + bias + resid   (output projection)
//   RESID=false: bf16 out = A@Bt^T + bias           (input projection)
// ---------------------------------------------------------------------------
template <bool RESID>
__global__ void gemm_bf16_2x2_kernel(const bf16_t* __restrict__ A,
                                     const bf16_t* __restrict__ Bt,
                                     const float* __restrict__ bias,
                                     const float* __restrict__ resid,
                                     bf16_t* __restrict__ outbf,
                                     float* __restrict__ outf,
                                     int M, int N, int K) {
  int lane = threadIdx.x & 31;
  int gw = blockIdx.x * (RT / 32) + (threadIdx.x >> 5);
  int ntn2 = N >> 5;                       // 32-wide N blocks
  int mt2 = gw / ntn2, nc2 = gw - mt2 * ntn2;
  if (mt2 >= (M >> 5)) return;             // uniform per wave
  int n0 = nc2 << 5;
  const bf16_t* arow0 = A + (size_t)(mt2 * 32 + (lane & 15)) * K;
  const bf16_t* arow1 = arow0 + (size_t)16 * K;
  v8f c00 = {}, c01 = {}, c10 = {}, c11 = {};
#pragma unroll 2
  for (int k0 = 0; k0 < K; k0 += 32) {
    v16bf a0 = load_a_frag(arow0, k0, lane);
    v16bf a1 = load_a_frag(arow1, k0, lane);
    v16bf b0 = load_b_frag(Bt, K, n0, k0, lane);
    v16bf b1 = load_b_frag(Bt, K, n0 + 16, k0, lane);
    c00 = wmma_bf16(a0, b0, c00);
    c01 = wmma_bf16(a0, b1, c01);
    c10 = wmma_bf16(a1, b0, c10);
    c11 = wmma_bf16(a1, b1, c11);
  }
  int crow = (lane >> 4) << 3;
  int ncol = lane & 15;
#pragma unroll
  for (int ti = 0; ti < 2; ++ti) {
#pragma unroll
    for (int tj = 0; tj < 2; ++tj) {
      const v8f& c = ti ? (tj ? c11 : c10) : (tj ? c01 : c00);
      int n = n0 + tj * 16 + ncol;
      float bv = bias[n];
#pragma unroll
      for (int r = 0; r < 8; ++r) {
        int m = mt2 * 32 + ti * 16 + crow + r;
        float v = c[r] + bv;
        if (RESID) outf[(size_t)m * N + n] = v + resid[(size_t)m * N + n];
        else       outbf[(size_t)m * N + n] = (bf16_t)v;
      }
    }
  }
}

// ---------------------------------------------------------------------------
// Persistent recurrent kernel
// ---------------------------------------------------------------------------
__device__ __forceinline__ void grid_barrier(unsigned* __restrict__ cnt,
                                             unsigned* __restrict__ flag, int idx) {
  __syncthreads();
  if (threadIdx.x == 0) {
    __threadfence();
    unsigned prev = __hip_atomic_fetch_add(&cnt[idx], 1u, __ATOMIC_ACQ_REL,
                                           __HIP_MEMORY_SCOPE_AGENT);
    if (prev == (unsigned)(RB - 1)) {
      __hip_atomic_store(&flag[idx], 1u, __ATOMIC_RELEASE, __HIP_MEMORY_SCOPE_AGENT);
    } else {
      while (!__hip_atomic_load(&flag[idx], __ATOMIC_ACQUIRE, __HIP_MEMORY_SCOPE_AGENT))
        __builtin_amdgcn_s_sleep(1);
    }
  }
  __syncthreads();
}

__global__ __launch_bounds__(RT) void cfc_recurrent_kernel(
    const bf16_t* __restrict__ cfcx,                 // [B][T][UNITS] bf16
    const bf16_t* __restrict__ Wbb0t,                // [BBU][BBU]
    const bf16_t* __restrict__ Wbb1t,                // [BBU][BBU]
    const bf16_t* __restrict__ Wheadt,               // [4*UNITS][BBU] (ff1|ff2|ta|tb)
    const float* __restrict__ bb_b,                  // [2][BBU]
    const float* __restrict__ b_ff1, const float* __restrict__ b_ff2,
    const float* __restrict__ b_ta,  const float* __restrict__ b_tb,
    const float* __restrict__ ts,                    // [B][T]
    bf16_t* __restrict__ hbf, float* __restrict__ hf32,   // [B][UNITS]
    bf16_t* __restrict__ act0, bf16_t* __restrict__ act1, // [B][BBU]
    bf16_t* __restrict__ cfcout,                     // [B][T][UNITS]
    unsigned* __restrict__ bcnt, unsigned* __restrict__ bflag) {
  const int lane = threadIdx.x & 31;
  const int gw = blockIdx.x * (RT / 32) + (threadIdx.x >> 5);  // 0..255
  const int arow = lane & 15;           // A-row within 16-row tile
  const int crow = (lane >> 4) << 3;    // C/D row base within tile

  for (int t = 0; t < T_SZ; ++t) {
    // ---- backbone layer 0: act0 = lecun_tanh([cfc_x_t | h] @ Wbb0 + b0) ----
    {
      int nc = gw & 127, mt = gw >> 7;
      int n0 = nc << 4;
      int b = mt * 16 + arow;
      const bf16_t* axr = cfcx + ((size_t)b * T_SZ + t) * UNITS;
      const bf16_t* ahr = hbf + (size_t)b * UNITS;
      v8f c = {};
      // Split K reduction: loop-invariant A base pointers, no per-chunk select.
#pragma unroll 4
      for (int k0 = 0; k0 < UNITS; k0 += 32)           // K in [0,1024): cfc_x_t
        c = wmma_bf16(load_a_frag(axr, k0, lane),
                      load_b_frag(Wbb0t, BBU, n0, k0, lane), c);
#pragma unroll 4
      for (int k0 = 0; k0 < UNITS; k0 += 32)           // K in [1024,2048): h
        c = wmma_bf16(load_a_frag(ahr, k0, lane),
                      load_b_frag(Wbb0t, BBU, n0, UNITS + k0, lane), c);
      int n = n0 + (lane & 15);
      float bv = bb_b[n];
#pragma unroll
      for (int r = 0; r < 8; ++r) {
        int m = mt * 16 + crow + r;
        act0[(size_t)m * BBU + n] = (bf16_t)lecun_tanh(c[r] + bv);
      }
    }
    grid_barrier(bcnt, bflag, t * 3 + 0);

    // ---- backbone layer 1: act1 = lecun_tanh(act0 @ Wbb1 + b1) ----
    {
      int nc = gw & 127, mt = gw >> 7;
      int n0 = nc << 4;
      const bf16_t* ar = act0 + (size_t)(mt * 16 + arow) * BBU;
      v8f c = {};
#pragma unroll 4
      for (int k0 = 0; k0 < BBU; k0 += 32)
        c = wmma_bf16(load_a_frag(ar, k0, lane),
                      load_b_frag(Wbb1t, BBU, n0, k0, lane), c);
      int n = n0 + (lane & 15);
      float bv = bb_b[BBU + n];
#pragma unroll
      for (int r = 0; r < 8; ++r) {
        int m = mt * 16 + crow + r;
        act1[(size_t)m * BBU + n] = (bf16_t)lecun_tanh(c[r] + bv);
      }
    }
    grid_barrier(bcnt, bflag, t * 3 + 1);

    // ---- heads + CfC gate: one wave owns the same (m,n) tile in all 4 heads
    if (gw < 128) {
      int nc = gw & 63, mt = gw >> 6;
      int n0 = nc << 4;
      const bf16_t* ar = act1 + (size_t)(mt * 16 + arow) * BBU;
      v8f c0 = {}, c1 = {}, c2 = {}, c3 = {};
#pragma unroll 2
      for (int k0 = 0; k0 < BBU; k0 += 32) {
        v16bf a = load_a_frag(ar, k0, lane);   // shared across 4 heads
        c0 = wmma_bf16(a, load_b_frag(Wheadt, BBU, n0,             k0, lane), c0);
        c1 = wmma_bf16(a, load_b_frag(Wheadt, BBU, n0 + UNITS,     k0, lane), c1);
        c2 = wmma_bf16(a, load_b_frag(Wheadt, BBU, n0 + 2 * UNITS, k0, lane), c2);
        c3 = wmma_bf16(a, load_b_frag(Wheadt, BBU, n0 + 3 * UNITS, k0, lane), c3);
      }
      int n = n0 + (lane & 15);
      float bf1 = b_ff1[n], bf2 = b_ff2[n], bta = b_ta[n], btb = b_tb[n];
#pragma unroll
      for (int r = 0; r < 8; ++r) {
        int m = mt * 16 + crow + r;                     // batch index
        float tsv = ts[(size_t)m * T_SZ + t];
        float f1 = fast_tanh(c0[r] + bf1);
        float f2 = fast_tanh(c1[r] + bf2);
        float ta = c2[r] + bta, tb = c3[r] + btb;
        float ti = 1.0f / (1.0f + __expf(-(ta * tsv + tb)));
        float hn = f1 * (1.0f - ti) + ti * f2;
        hbf[(size_t)m * UNITS + n] = (bf16_t)hn;        // next-step A operand
        hf32[(size_t)m * UNITS + n] = hn;               // exact h_final
        cfcout[((size_t)m * T_SZ + t) * UNITS + n] = (bf16_t)hn;
      }
    }
    grid_barrier(bcnt, bflag, t * 3 + 2);
  }
}

// ---------------------------------------------------------------------------
// Host launcher
// ---------------------------------------------------------------------------
extern "C" void kernel_launch(void* const* d_in, const int* in_sizes, int n_in,
                              void* d_out, int out_size, void* d_ws, size_t ws_size,
                              hipStream_t stream) {
  (void)in_sizes; (void)n_in; (void)out_size; (void)ws_size;
  const float* x      = (const float*)d_in[0];
  const float* ts     = (const float*)d_in[1];
  const float* hidden = (const float*)d_in[2];
  const float* norm_w = (const float*)d_in[3];
  const float* norm_b = (const float*)d_in[4];
  const float* w_in   = (const float*)d_in[5];
  const float* b_in   = (const float*)d_in[6];
  const float* bb_w   = (const float*)d_in[7];
  const float* bb_b   = (const float*)d_in[8];
  const float* w_ff1  = (const float*)d_in[9];
  const float* b_ff1  = (const float*)d_in[10];
  const float* w_ff2  = (const float*)d_in[11];
  const float* b_ff2  = (const float*)d_in[12];
  const float* w_ta   = (const float*)d_in[13];
  const float* b_ta   = (const float*)d_in[14];
  const float* w_tb   = (const float*)d_in[15];
  const float* b_tb   = (const float*)d_in[16];
  const float* w_out  = (const float*)d_in[17];
  const float* b_out  = (const float*)d_in[18];

  const size_t MROWS = (size_t)B_SZ * T_SZ;  // 16384

  uint8_t* ws = (uint8_t*)d_ws;
  size_t off = 0;
  auto carve = [&](size_t bytes) -> void* {
    void* p = ws + off;
    off = (off + bytes + 255) & ~(size_t)255;
    return p;
  };
  bf16_t* Wint   = (bf16_t*)carve((size_t)DM * DM * 2);
  bf16_t* Woutt  = (bf16_t*)carve((size_t)UNITS * DM * 2);
  bf16_t* Wbb0t  = (bf16_t*)carve((size_t)BBU * BBU * 2);
  bf16_t* Wbb1t  = (bf16_t*)carve((size_t)BBU * BBU * 2);
  bf16_t* Wheadt = (bf16_t*)carve((size_t)4 * UNITS * BBU * 2);
  bf16_t* xn     = (bf16_t*)carve(MROWS * DM * 2);
  bf16_t* cfcx   = (bf16_t*)carve(MROWS * UNITS * 2);
  bf16_t* cfcout = (bf16_t*)carve(MROWS * UNITS * 2);
  bf16_t* act0   = (bf16_t*)carve((size_t)B_SZ * BBU * 2);
  bf16_t* act1   = (bf16_t*)carve((size_t)B_SZ * BBU * 2);
  bf16_t* hbf    = (bf16_t*)carve((size_t)B_SZ * UNITS * 2);
  float*  hf32   = (float*)carve((size_t)B_SZ * UNITS * 4);
  unsigned* bcnt  = (unsigned*)carve((size_t)NBAR * 4);
  unsigned* bflag = (unsigned*)carve((size_t)NBAR * 4);

  // 1. state / barrier init (fresh every call -> graph-replay deterministic)
  zero_u32_kernel<<<(2 * NBAR + 255) / 256, 256, 0, stream>>>(bcnt, 2 * NBAR);
  init_hidden_kernel<<<(B_SZ * UNITS + 255) / 256, 256, 0, stream>>>(
      hidden, hbf, hf32, B_SZ * UNITS);

  // 2. weight conversion: [K][N] fp32 -> [N][K] bf16
  transpose_bf16_kernel<<<2048, 256, 0, stream>>>(w_in, Wint, DM, DM);
  transpose_bf16_kernel<<<2048, 256, 0, stream>>>(w_out, Woutt, UNITS, DM);
  transpose_bf16_kernel<<<4096, 256, 0, stream>>>(bb_w, Wbb0t, BBU, BBU);
  transpose_bf16_kernel<<<4096, 256, 0, stream>>>(bb_w + (size_t)BBU * BBU, Wbb1t, BBU, BBU);
  transpose_bf16_kernel<<<2048, 256, 0, stream>>>(w_ff1, Wheadt + (size_t)0 * UNITS * BBU, BBU, UNITS);
  transpose_bf16_kernel<<<2048, 256, 0, stream>>>(w_ff2, Wheadt + (size_t)1 * UNITS * BBU, BBU, UNITS);
  transpose_bf16_kernel<<<2048, 256, 0, stream>>>(w_ta,  Wheadt + (size_t)2 * UNITS * BBU, BBU, UNITS);
  transpose_bf16_kernel<<<2048, 256, 0, stream>>>(w_tb,  Wheadt + (size_t)3 * UNITS * BBU, BBU, UNITS);

  // 3. LayerNorm
  layernorm_bf16_kernel<<<(int)MROWS, 256, 0, stream>>>(x, norm_w, norm_b, xn);

  // 4. input projection: cfc_x = xn @ w_in + b_in  (bf16 out)
  {
    int tiles = (int)((MROWS / 32) * (DM / 32));  // 16384 waves
    gemm_bf16_2x2_kernel<false><<<tiles / (RT / 32), RT, 0, stream>>>(
        xn, Wint, b_in, nullptr, cfcx, nullptr, (int)MROWS, DM, DM);
  }

  // 5. persistent recurrence (512 steps, grid-wide barriers)
  cfc_recurrent_kernel<<<RB, RT, 0, stream>>>(
      cfcx, Wbb0t, Wbb1t, Wheadt, bb_b, b_ff1, b_ff2, b_ta, b_tb, ts,
      hbf, hf32, act0, act1, cfcout, bcnt, bflag);

  // 6. output projection + residual: y = x + cfc_out @ w_out + b_out (fp32)
  {
    int tiles = (int)((MROWS / 32) * (DM / 32));
    gemm_bf16_2x2_kernel<true><<<tiles / (RT / 32), RT, 0, stream>>>(
        cfcout, Woutt, b_out, x, nullptr, (float*)d_out, (int)MROWS, UNITS, DM);
  }

  // 7. h_final (fp32, exact) appended after y
  copy_f32_kernel<<<(B_SZ * UNITS + 255) / 256, 256, 0, stream>>>(
      hf32, (float*)d_out + MROWS * DM, B_SZ * UNITS);
}
